// SIRENSparse_46952582479973
// MI455X (gfx1250) — compile-verified, matrix operand order checked by reading, not run on hardware
//
#include <hip/hip_runtime.h>

typedef __attribute__((ext_vector_type(16))) _Float16 v16h;
typedef __attribute__((ext_vector_type(8)))  _Float16 v8h;
typedef __attribute__((ext_vector_type(4)))  _Float16 v4h;
typedef __attribute__((ext_vector_type(8)))  float    v8f;
typedef __attribute__((ext_vector_type(4)))  unsigned int v4u;
typedef __attribute__((ext_vector_type(8)))  int      v8i;
typedef __attribute__((ext_vector_type(4)))  int      v4i;

#define WIDTH   256
#define ROWS    128          // points per workgroup
#define W0FIRST 30.0f

#define ACT_HALVES   (ROWS * WIDTH)            // 32768 halves = 64 KB
#define WMAT_HALVES  (WIDTH * WIDTH)           // 65536 halves = 128 KB
#define WS_BYTES_NEEDED (4u * WMAT_HALVES * sizeof(_Float16))   // 512 KB

// ---------------------------------------------------------------------------
// TDM: one DMA moves a full 256x256 f16 weight matrix (128 KB) global -> LDS.
// D# built per CDNA5 ISA (2D tensor, data_size=8B units,
// tensor_dim0=stride=64 (512B rows), tile = 64 x 256 = whole matrix, type=2).
// ---------------------------------------------------------------------------
__device__ __forceinline__ void tdm_load_weights(const _Float16* gsrc,
                                                 unsigned lds_byte_off) {
    unsigned long long ga = (unsigned long long)gsrc;
    v4u g0 = { 0x1u,                                   // count=1, user-mode
               lds_byte_off,                           // lds_addr (bytes)
               (unsigned)(ga & 0xFFFFFFFFu),           // global_addr[31:0]
               (unsigned)((ga >> 32) & 0x1FFFFFFu) | 0x80000000u }; // [56:32] | type=2
    v8i g1 = { 0x00030000,          // wg_mask=0, data_size=3 (8-byte units)
               64 << 16,            // tensor_dim0 = 64 (bits 63:48)
               256 << 16,           // tensor_dim1 = 256 (bits 95:80)
               64 << 16,            // tile_dim0 = 64 (bits 127:112)
               256,                 // tile_dim1 = 256, tile_dim2 = 0
               64,                  // tensor_dim0_stride = 64
               0, 0 };
    v4i z4 = {0, 0, 0, 0};
#if __clang_major__ >= 23
    v8i z8 = {0, 0, 0, 0, 0, 0, 0, 0};
    __builtin_amdgcn_tensor_load_to_lds(g0, g1, z4, z4, z8, 0);
#else
    __builtin_amdgcn_tensor_load_to_lds(g0, g1, z4, z4, 0);
#endif
}

// ---------------------------------------------------------------------------
// One hidden layer for this wave's 16 rows: A fragments held in VGPRs across
// all 16 column tiles; 2 column tiles per iteration (2 accumulators) for ILP.
// ---------------------------------------------------------------------------
__device__ __forceinline__ void hidden_layer(_Float16* __restrict__ act,
                                             const _Float16* __restrict__ wl,
                                             const float* __restrict__ bg,
                                             int wv, int lrow, int sel) {
    // 16-bit A 16x32 layout: lanes 0-15 hold K[0..7]+K[16..23],
    // lanes 16-31 hold K[8..15]+K[24..31]  (two ds_load_b128 per fragment).
    const _Float16* arow = act + (wv * 16 + lrow) * WIDTH;
    v16h afrag[8];
#pragma unroll
    for (int kc = 0; kc < 8; ++kc) {
        union { v16h v; v8h h[2]; } u;
        u.h[0] = *(const v8h*)(arow + kc * 32 + sel * 8);
        u.h[1] = *(const v8h*)(arow + kc * 32 + 16 + sel * 8);
        afrag[kc] = u.v;
    }
    for (int ct = 0; ct < 16; ct += 2) {
        const int col0 = ct * 16 + lrow;
        const int col1 = col0 + 16;
        const _Float16* wrow0 = wl + col0 * WIDTH;   // W[col][k], contiguous in k
        const _Float16* wrow1 = wl + col1 * WIDTH;
        v8f c0 = {0.f, 0.f, 0.f, 0.f, 0.f, 0.f, 0.f, 0.f};
        v8f c1 = {0.f, 0.f, 0.f, 0.f, 0.f, 0.f, 0.f, 0.f};
#pragma unroll
        for (int kc = 0; kc < 8; ++kc) {
            union { v16h v; v8h h[2]; } b0, b1;      // B lane = out col, K like A
            b0.h[0] = *(const v8h*)(wrow0 + kc * 32 + sel * 8);
            b0.h[1] = *(const v8h*)(wrow0 + kc * 32 + 16 + sel * 8);
            b1.h[0] = *(const v8h*)(wrow1 + kc * 32 + sel * 8);
            b1.h[1] = *(const v8h*)(wrow1 + kc * 32 + 16 + sel * 8);
            c0 = __builtin_amdgcn_wmma_f32_16x16x32_f16(
                     false, afrag[kc], false, b0.v, (short)0, c0, false, false);
            c1 = __builtin_amdgcn_wmma_f32_16x16x32_f16(
                     false, afrag[kc], false, b1.v, (short)0, c1, false, false);
        }
        // D layout: VGPR r -> row M = r + 8*sel, lane -> col. sin + f16 in place.
        const float bias0 = bg[col0];
        const float bias1 = bg[col1];
#pragma unroll
        for (int r = 0; r < 8; ++r) {
            const int rowOut = wv * 16 + sel * 8 + r;
            act[rowOut * WIDTH + col0] = (_Float16)__sinf(c0[r] + bias0);
            act[rowOut * WIDTH + col1] = (_Float16)__sinf(c1[r] + bias1);
        }
    }
}

__device__ __forceinline__ void first_layer(_Float16* __restrict__ act,
                                            const float* __restrict__ xyt,
                                            const float* __restrict__ W0,
                                            const float* __restrict__ b0,
                                            int tid, long row0, int N) {
    const int col = tid;  // blockDim == WIDTH
    const float w00 = W0[col * 3 + 0];
    const float w01 = W0[col * 3 + 1];
    const float w02 = W0[col * 3 + 2];
    const float bz  = b0[col];
    for (int r = 0; r < ROWS; ++r) {
        long row = row0 + r;
        row = row < N ? row : (long)(N - 1);
        const float x0 = xyt[row * 3 + 0];
        const float x1 = xyt[row * 3 + 1];
        const float x2 = xyt[row * 3 + 2];
        const float pre = W0FIRST * fmaf(x0, w00, fmaf(x1, w01, fmaf(x2, w02, bz)));
        act[r * WIDTH + col] = (_Float16)__sinf(pre);
    }
}

__device__ __forceinline__ void final_layer(const _Float16* __restrict__ act,
                                            const float* __restrict__ Wf,
                                            const float* __restrict__ bf,
                                            float* __restrict__ out,
                                            int tid, long row0, int N) {
    if (tid < ROWS) {
        const _Float16* h = act + tid * WIDTH;
        float s = bf[0];
        for (int k = 0; k < WIDTH; k += 8) {
            const v8h hv = *(const v8h*)(h + k);
#pragma unroll
            for (int j = 0; j < 8; ++j)
                s = fmaf((float)hv[j], Wf[k + j], s);
        }
        const long row = row0 + tid;
        if (row < N) out[row] = s;
    }
}

// ---------------------------------------------------------------------------
// Prep: convert 4 hidden-layer weight matrices f32 -> f16 into d_ws
// (layout unchanged: [layer][out_col][k]).
// ---------------------------------------------------------------------------
__global__ __launch_bounds__(256)
void siren_prep_kernel(const float* __restrict__ W1, const float* __restrict__ W2,
                       const float* __restrict__ W3, const float* __restrict__ W4,
                       _Float16* __restrict__ wsw) {
    const float* Ws[4] = {W1, W2, W3, W4};
    const unsigned gid = blockIdx.x * 256 + threadIdx.x;   // one float4 each
    if (gid < 4u * WMAT_HALVES / 4u) {
        const int l   = gid >> 14;                         // 16384 float4 / layer
        const int off = gid & 16383;
        const float4 w = ((const float4*)Ws[l])[off];
        v4h h4 = {(_Float16)w.x, (_Float16)w.y, (_Float16)w.z, (_Float16)w.w};
        *(v4h*)(wsw + 4u * gid) = h4;
    }
}

// ---------------------------------------------------------------------------
// Main kernel (TDM path): act 64KB + double-buffered weights 2x128KB = 320KB.
// TDM fetch of layer l+1 overlaps WMMA compute of layer l.
// ---------------------------------------------------------------------------
__global__ __launch_bounds__(256)
void siren_wmma_tdm_kernel(const float* __restrict__ xyt,
                           const float* __restrict__ W0, const float* __restrict__ b0,
                           const float* __restrict__ b1, const float* __restrict__ b2,
                           const float* __restrict__ b3, const float* __restrict__ b4,
                           const float* __restrict__ Wf, const float* __restrict__ bf,
                           const _Float16* __restrict__ wsw,   // pre-converted f16 weights
                           float* __restrict__ out, int N)
{
    extern __shared__ _Float16 smem[];
    _Float16* act = smem;                                   // 64 KB

    const int  tid  = threadIdx.x;
    const long row0 = (long)blockIdx.x * ROWS;
    const int lane = tid & 31;
    const int wv   = tid >> 5;
    const int lrow = lane & 15;
    const int sel  = lane >> 4;

    // Kick off layer-1 weight DMA immediately; overlap with scalar first layer.
    if (wv == 0)
        tdm_load_weights(wsw, (unsigned)(ACT_HALVES * sizeof(_Float16)));

    first_layer(act, xyt, W0, b0, tid, row0, N);

    const float* bs[4] = {b1, b2, b3, b4};
    for (int l = 0; l < 4; ++l) {
        __syncthreads();   // all waves done reading buffer (l+1)&1 from layer l-1
        if (wv == 0) {
            if (l < 3) {   // prefetch next layer's weights into the other buffer
                const unsigned dst = (unsigned)((ACT_HALVES +
                                     ((l + 1) & 1) * WMAT_HALVES) * sizeof(_Float16));
                tdm_load_weights(wsw + (l + 1) * WMAT_HALVES, dst);
                __builtin_amdgcn_s_wait_tensorcnt(1);  // layer l's DMA complete
            } else {
                __builtin_amdgcn_s_wait_tensorcnt(0);
            }
        }
        __syncthreads();   // weights for layer l visible to all waves
        // double-buffer base via offset arithmetic (no LDS pointer arrays!)
        const _Float16* wl = smem + ACT_HALVES + (unsigned)(l & 1) * WMAT_HALVES;
        hidden_layer(act, wl, bs[l], wv, lrow, sel);
    }

    __syncthreads();
    final_layer(act, Wf, bf, out, tid, row0, N);
}

// ---------------------------------------------------------------------------
// Fallback (no workspace): in-kernel f32->f16 staging, 192 KB LDS.
// ---------------------------------------------------------------------------
__global__ __launch_bounds__(256)
void siren_wmma_fallback(const float* __restrict__ xyt,
                         const float* __restrict__ W0, const float* __restrict__ b0,
                         const float* __restrict__ W1, const float* __restrict__ b1,
                         const float* __restrict__ W2, const float* __restrict__ b2,
                         const float* __restrict__ W3, const float* __restrict__ b3,
                         const float* __restrict__ W4, const float* __restrict__ b4,
                         const float* __restrict__ Wf, const float* __restrict__ bf,
                         float* __restrict__ out, int N)
{
    extern __shared__ _Float16 smem[];
    _Float16* act = smem;
    _Float16* wl  = smem + ACT_HALVES;

    const int  tid  = threadIdx.x;
    const long row0 = (long)blockIdx.x * ROWS;
    const int lane = tid & 31;
    const int wv   = tid >> 5;
    const int lrow = lane & 15;
    const int sel  = lane >> 4;

    first_layer(act, xyt, W0, b0, tid, row0, N);

    const float* Ws[4] = {W1, W2, W3, W4};
    const float* bs[4] = {b1, b2, b3, b4};
    for (int l = 0; l < 4; ++l) {
        __syncthreads();
        const float* Wg = Ws[l];
        for (int i = tid; i < (WIDTH * WIDTH) / 4; i += 256) {
            const float4 w = ((const float4*)Wg)[i];
            v4h h4 = {(_Float16)w.x, (_Float16)w.y, (_Float16)w.z, (_Float16)w.w};
            *(v4h*)(wl + 4 * i) = h4;
        }
        __syncthreads();
        hidden_layer(act, wl, bs[l], wv, lrow, sel);
    }

    __syncthreads();
    final_layer(act, Wf, bf, out, tid, row0, N);
}

extern "C" void kernel_launch(void* const* d_in, const int* in_sizes, int n_in,
                              void* d_out, int out_size, void* d_ws, size_t ws_size,
                              hipStream_t stream) {
    const float* xyt = (const float*)d_in[0];
    const float* W0  = (const float*)d_in[1];
    const float* b0  = (const float*)d_in[2];
    const float* W1  = (const float*)d_in[3];
    const float* b1  = (const float*)d_in[4];
    const float* W2  = (const float*)d_in[5];
    const float* b2  = (const float*)d_in[6];
    const float* W3  = (const float*)d_in[7];
    const float* b3  = (const float*)d_in[8];
    const float* W4  = (const float*)d_in[9];
    const float* b4  = (const float*)d_in[10];
    const float* Wf  = (const float*)d_in[11];
    const float* bf  = (const float*)d_in[12];
    float* out = (float*)d_out;

    const int N = in_sizes[0] / 3;
    const int blocks = (N + ROWS - 1) / ROWS;

    if (ws_size >= WS_BYTES_NEEDED) {
        _Float16* wsw = (_Float16*)d_ws;
        siren_prep_kernel<<<(4 * WMAT_HALVES / 4 + 255) / 256, 256, 0, stream>>>(
            W1, W2, W3, W4, wsw);

        const size_t shmem = (size_t)(ACT_HALVES + 2 * WMAT_HALVES) * sizeof(_Float16); // 320 KB
        (void)hipFuncSetAttribute((const void*)siren_wmma_tdm_kernel,
                                  hipFuncAttributeMaxDynamicSharedMemorySize, (int)shmem);
        siren_wmma_tdm_kernel<<<blocks, 256, shmem, stream>>>(
            xyt, W0, b0, b1, b2, b3, b4, Wf, bf, wsw, out, N);
    } else {
        const size_t shmem = (size_t)(ACT_HALVES + WMAT_HALVES) * sizeof(_Float16);     // 192 KB
        (void)hipFuncSetAttribute((const void*)siren_wmma_fallback,
                                  hipFuncAttributeMaxDynamicSharedMemorySize, (int)shmem);
        siren_wmma_fallback<<<blocks, 256, shmem, stream>>>(
            xyt, W0, b0, W1, b1, W2, b2, W3, b3, W4, b4, Wf, bf, out, N);
    }
}
